// Attention_28647431865068
// MI455X (gfx1250) — matrix operand order repaired and checked by reading, not compiled, not run on hardware
//
#include <hip/hip_runtime.h>
#include <hip/hip_bf16.h>

typedef __attribute__((ext_vector_type(16))) _Float16 v16h;
typedef __attribute__((ext_vector_type(8)))  _Float16 v8h;
typedef __attribute__((ext_vector_type(8)))  float    v8f;

#define BB 8
#define TT 2048
#define DD 1024
#define HH 128

// ---------------------------------------------------------------------------
// Pass 0: convert + transpose weights:  W[D][H] f32  ->  WT[H][D] f16
// (one-shot, 1.5 MB read; makes every B-fragment a single 32B v16h load)
// ---------------------------------------------------------------------------
__global__ __launch_bounds__(256)
void wconv_kernel(const float* __restrict__ Wq,
                  const float* __restrict__ Wk,
                  const float* __restrict__ Wv,
                  _Float16* __restrict__ WT) {       // [3][HH][DD]
  const int idx   = blockIdx.x * 256 + threadIdx.x;  // 3*HH*DD total
  const int which = idx / (HH * DD);
  const int col   = (idx / DD) % HH;
  const int k     = idx % DD;
  const float* W  = (which == 0) ? Wq : (which == 1) ? Wk : Wv;
  WT[idx] = (_Float16)W[(size_t)k * HH + col];
}

// ---------------------------------------------------------------------------
// Pass 1: Q/K/V projection.  One 256-thread block (8 waves) per 16-row tile.
// x tile (16x1024 fp32) is staged once into LDS as f16; wave w owns columns
// [w*16, w*16+16).  Single k-loop with THREE live accumulators so each A
// fragment is loaded from LDS exactly once and feeds three WMMAs (three
// independent accumulation chains also hide XDL latency).
// Fragment layouts used by this code (contiguous k, not the ISA table order) (wave32):
//   A (16x32 f16): lane m = lane%16 is the row; element e maps to
//                  k = e + (lane/16)*16
//   B (32x16 f16): lane n = lane%16 is the col; element e maps to
//                  k = e + (lane/16)*16
//   C/D (16x16 f32): lanes 0-15 N=lane M=r, lanes 16-31 N=lane-16 M=r+8
// ---------------------------------------------------------------------------
__global__ __launch_bounds__(256)
void qkv_proj_kernel(const float* __restrict__ x,
                     const _Float16* __restrict__ WT,   // [3][HH][DD] f16
                     _Float16* __restrict__ qbuf,       // [B*T][HH] (pre-scaled)
                     _Float16* __restrict__ kbuf,       // [B*T][HH]
                     _Float16* __restrict__ vtbuf) {    // [B][HH][TT]
  __shared__ __align__(32) _Float16 xs[16 * DD];        // 32 KB

  const int tid  = threadIdx.x;
  const int row0 = blockIdx.x * 16;                     // global row in [0, B*T)

  // cooperative x tile load: fp32 -> f16, coalesced
#pragma unroll
  for (int i = 0; i < 64; ++i) {
    const int idx = i * 256 + tid;                      // 16384 elements
    const int r   = idx >> 10;
    const int c   = idx & (DD - 1);
    xs[idx] = (_Float16)x[(size_t)(row0 + r) * DD + c];
  }
  __syncthreads();

  const int wave = tid >> 5;
  const int lane = tid & 31;
  const int l16  = lane & 15;
  const int hgrp = lane >> 4;
  const int ncol = wave * 16 + l16;
  const _Float16* arow = xs + l16 * DD;                 // A rows live in LDS

  const _Float16* wtq = WT + ((size_t)0 * HH + ncol) * DD;
  const _Float16* wtk = WT + ((size_t)1 * HH + ncol) * DD;
  const _Float16* wtv = WT + ((size_t)2 * HH + ncol) * DD;

  v8f cq = {}, ck = {}, cv = {};
  for (int k0 = 0; k0 < DD; k0 += 32) {
    v16h a;
    {
      a = *(const v16h*)(arow + k0 + hgrp * 16);
    }
    v16h bq = *(const v16h*)(wtq + k0 + hgrp * 16);          // 32B global loads
    v16h bk = *(const v16h*)(wtk + k0 + hgrp * 16);
    v16h bv = *(const v16h*)(wtv + k0 + hgrp * 16);
    cq = __builtin_amdgcn_wmma_f32_16x16x32_f16(false, a, false, bq,
                                                (short)0, cq, false, false);
    ck = __builtin_amdgcn_wmma_f32_16x16x32_f16(false, a, false, bk,
                                                (short)0, ck, false, false);
    cv = __builtin_amdgcn_wmma_f32_16x16x32_f16(false, a, false, bv,
                                                (short)0, cv, false, false);
  }

  // stores: Q pre-scaled by 1/sqrt(H); V stored transposed vt[b][col][t]
#pragma unroll
  for (int r = 0; r < 8; ++r) {
    const int row = row0 + r + hgrp * 8;
    qbuf[(size_t)row * HH + ncol] = (_Float16)(cq[r] * 0.08838834764831845f);
    kbuf[(size_t)row * HH + ncol] = (_Float16)ck[r];
    const int bb = row / TT, t = row % TT;
    vtbuf[((size_t)bb * HH + ncol) * TT + t] = (_Float16)cv[r];
  }
}

// ---------------------------------------------------------------------------
// Pass 2: causal flash attention.  One wave32 owns 16 query rows and the full
// H=128 output, streaming key blocks of 32 (so P is exactly one 16x32 f16
// A-fragment for the P*V WMMA).  Online softmax with running max / sum.
// ---------------------------------------------------------------------------
__global__ __launch_bounds__(32)
void flash_attn_kernel(const _Float16* __restrict__ qbuf,
                       const _Float16* __restrict__ kbuf,
                       const _Float16* __restrict__ vtbuf,
                       float* __restrict__ out) {
  __shared__ __align__(32) _Float16 pshm[16 * 32];  // P tile bounce buffer

  const int lane = threadIdx.x & 31;
  const int l16  = lane & 15;
  const int hgrp = lane >> 4;
  const int b    = blockIdx.y;
  const int q0   = blockIdx.x * 16;

  const _Float16* qb = qbuf  + (size_t)b * TT * HH;
  const _Float16* kq = kbuf  + (size_t)b * TT * HH;
  const _Float16* vt = vtbuf + (size_t)b * HH * TT;

  // Q fragments for all 4 K-steps over H, kept in registers
  v16h qf[4];
  {
    const _Float16* qrow = qb + (size_t)(q0 + l16) * HH;
#pragma unroll
    for (int hs = 0; hs < 4; ++hs) {
      qf[hs] = *(const v16h*)(qrow + hs * 32 + hgrp * 16);
    }
  }

  v8f o[8];
#pragma unroll
  for (int t = 0; t < 8; ++t) o[t] = (v8f){};
  float mrow[8], lrow[8];
#pragma unroll
  for (int r = 0; r < 8; ++r) { mrow[r] = -1e30f; lrow[r] = 0.0f; }

  const int nkb = (q0 + 47) / 32;  // causal: key blocks covering keys <= q0+15
  for (int kbk = 0; kbk < nkb; ++kbk) {
    const int key0 = kbk * 32;

    // prefetch next key block (K rows + V^T columns) -> global_prefetch_b8
    if (kbk + 1 < nkb) {
      const int nk = key0 + 32;
      __builtin_prefetch(kq + (size_t)(nk + lane) * HH, 0, 1);
      __builtin_prefetch(kq + (size_t)(nk + lane) * HH + 64, 0, 1);
#pragma unroll
      for (int c4 = 0; c4 < 4; ++c4)
        __builtin_prefetch(vt + (size_t)(lane * 4 + c4) * TT + nk, 0, 1);
    }

    // S = Q * K^T for 32 keys (two 16x16 tiles), K-loop over H
    v8f c0 = (v8f){}, c1 = (v8f){};
#pragma unroll
    for (int hs = 0; hs < 4; ++hs) {
      v16h kf0 = *(const v16h*)(kq + (size_t)(key0 + l16) * HH + hs * 32 + hgrp * 16);
      v16h kf1 = *(const v16h*)(kq + (size_t)(key0 + 16 + l16) * HH + hs * 32 + hgrp * 16);
      c0 = __builtin_amdgcn_wmma_f32_16x16x32_f16(false, qf[hs], false, kf0,
                                                  (short)0, c0, false, false);
      c1 = __builtin_amdgcn_wmma_f32_16x16x32_f16(false, qf[hs], false, kf1,
                                                  (short)0, c1, false, false);
    }

    // causal mask + online softmax; row = q0 + r + hgrp*8, cols across 16 lanes
#pragma unroll
    for (int r = 0; r < 8; ++r) {
      const int row  = q0 + r + hgrp * 8;
      const int keyA = key0 + l16;
      const int keyB = keyA + 16;
      float s0 = (keyA <= row) ? c0[r] : -1e30f;
      float s1 = (keyB <= row) ? c1[r] : -1e30f;

      float v = fmaxf(s0, s1);
#pragma unroll
      for (int mask = 1; mask < 16; mask <<= 1)
        v = fmaxf(v, __shfl_xor(v, mask, 32));
      const float mnew  = fmaxf(mrow[r], v);
      const float alpha = __expf(mrow[r] - mnew);
      const float p0 = __expf(s0 - mnew);
      const float p1 = __expf(s1 - mnew);
      float sum = p0 + p1;
#pragma unroll
      for (int mask = 1; mask < 16; mask <<= 1)
        sum += __shfl_xor(sum, mask, 32);
      lrow[r] = lrow[r] * alpha + sum;
      mrow[r] = mnew;
#pragma unroll
      for (int t = 0; t < 8; ++t) o[t][r] *= alpha;

      // stash P (C layout) to LDS for re-layout into an A fragment
      pshm[(r + hgrp * 8) * 32 + l16]      = (_Float16)p0;
      pshm[(r + hgrp * 8) * 32 + 16 + l16] = (_Float16)p1;
    }
    __syncthreads();

    // reload P as a 16x32 f16 A-fragment
    v16h pa;
    {
      const _Float16* pr = pshm + l16 * 32;
      pa = *(const v16h*)(pr + hgrp * 16);
    }
    __syncthreads();

    // O += P * V: V^T layout makes each B-fragment one 32B v16h load
#pragma unroll
    for (int t = 0; t < 8; ++t) {
      v16h vf = *(const v16h*)(vt + (size_t)(t * 16 + l16) * TT + key0 + hgrp * 16);
      o[t] = __builtin_amdgcn_wmma_f32_16x16x32_f16(false, pa, false, vf,
                                                    (short)0, o[t], false, false);
    }
  }

  // epilogue: normalize by running sum and store fp32
#pragma unroll
  for (int r = 0; r < 8; ++r) {
    const int row = q0 + r + hgrp * 8;
    const float inv = 1.0f / lrow[r];
    float* orow = out + ((size_t)b * TT + row) * HH;
#pragma unroll
    for (int t = 0; t < 8; ++t)
      orow[t * 16 + l16] = o[t][r] * inv;
  }
}

// ---------------------------------------------------------------------------
extern "C" void kernel_launch(void* const* d_in, const int* in_sizes, int n_in,
                              void* d_out, int out_size, void* d_ws, size_t ws_size,
                              hipStream_t stream) {
  (void)in_sizes; (void)n_in; (void)out_size; (void)ws_size;
  const float* x  = (const float*)d_in[0];
  const float* Wq = (const float*)d_in[1];
  const float* Wk = (const float*)d_in[2];
  const float* Wv = (const float*)d_in[3];
  float* out = (float*)d_out;

  const size_t n = (size_t)BB * TT * HH;   // elements per Q/K/V buffer
  _Float16* qbuf  = (_Float16*)d_ws;       // 4 MB
  _Float16* kbuf  = qbuf + n;              // 4 MB
  _Float16* vtbuf = kbuf + n;              // 4 MB (transposed V)
  _Float16* WT    = vtbuf + n;             // 768 KB (f16 W^T, [3][H][D])

  wconv_kernel<<<(3 * HH * DD) / 256, 256, 0, stream>>>(Wq, Wk, Wv, WT);

  qkv_proj_kernel<<<BB * TT / 16, 256, 0, stream>>>(x, WT, qbuf, kbuf, vtbuf);

  dim3 gAttn(TT / 16, BB);                 // (128, 8), 1 wave per block
  flash_attn_kernel<<<gAttn, 32, 0, stream>>>(qbuf, kbuf, vtbuf, out);
}
